// CvT_Block_7550552506683
// MI455X (gfx1250) — compile-verified, matrix operand order checked by reading, not run on hardware
//
#include <hip/hip_runtime.h>
#include <hip/hip_bf16.h>

// Problem constants
constexpr int kB  = 32;
constexpr int kC  = 128;
constexpr int kL  = 32;
constexpr int kH  = 8;
constexpr int kDK = 64;
constexpr int kN  = kL * kL;        // 1024 tokens per image
constexpr int kHD = kH * kDK;       // 512
constexpr float kInvLayer  = 0.44721359549995793f;  // 1/sqrt(NLAYER+1)
constexpr float kInvSqrtDk = 0.125f;                // 1/sqrt(64)

typedef __attribute__((ext_vector_type(16))) _Float16 v16h;
typedef __attribute__((ext_vector_type(8)))  _Float16 v8h;
typedef __attribute__((ext_vector_type(8)))  float    v8f;
typedef __attribute__((ext_vector_type(4)))  int      v4i;

// gfx1250 async global->LDS copy support (guarded: falls back to sync copy)
#if defined(__AMDGCN__) && __has_builtin(__builtin_amdgcn_global_load_async_to_lds_b128)
#define HAS_ASYNC_LDS 1
#else
#define HAS_ASYNC_LDS 0
#endif

#if HAS_ASYNC_LDS
typedef __attribute__((address_space(1))) v4i* gv4i_p;
typedef __attribute__((address_space(3))) v4i* lv4i_p;
#endif

__device__ inline void async_copy16(const _Float16* g, _Float16* l) {
#if HAS_ASYNC_LDS
  __builtin_amdgcn_global_load_async_to_lds_b128(
      (gv4i_p)(v4i*)g, (lv4i_p)(v4i*)l, /*offset=*/0, /*cpol=*/0);
#else
  *(v8h*)l = *(const v8h*)g;
#endif
}
__device__ inline void async_wait0() {
#if HAS_ASYNC_LDS
  asm volatile("s_wait_asynccnt 0" ::: "memory");
#endif
}

// ---------------------------------------------------------------------------
// WMMA helpers (gfx1250, wave32). 16x16x32 f16 -> f32.
// A-matrix 16x32 f16 layout (ISA 7.12.2): lane l holds row (l&15);
//   a[0..7]  = k {0..7}(lanes 0-15)/{8..15}(lanes 16-31)
//   a[8..15] = k {16..23}/{24..31}
// B-matrix mirrors A with lane = column. Works for global or LDS pointers.
// ---------------------------------------------------------------------------
__device__ inline v8f wmma16(v16h a, v16h b, v8f c) {
  return __builtin_amdgcn_wmma_f32_16x16x32_f16(
      false, a, false, b, (short)0, c, false, false);
}

__device__ inline v16h load_tile_A(const _Float16* p, int ld) {
  int lane = threadIdx.x & 31;
  int row  = lane & 15;
  int kh   = (lane >> 4) * 8;
  const _Float16* r = p + (size_t)row * ld + kh;
  v8h lo = *(const v8h*)(r);
  v8h hi = *(const v8h*)(r + 16);
  return __builtin_shufflevector(lo, hi, 0, 1, 2, 3, 4, 5, 6, 7,
                                 8, 9, 10, 11, 12, 13, 14, 15);
}

// C/D f32 layout: VGPR r -> (row = (lane>>4)*8 + r, col = lane&15)
__device__ inline void store_tile_D_f16(v8f d, _Float16* p, int ld) {
  int lane = threadIdx.x & 31;
  int col  = lane & 15;
  int row0 = (lane >> 4) * 8;
#pragma unroll
  for (int r = 0; r < 8; ++r) p[(size_t)(row0 + r) * ld + col] = (_Float16)d[r];
}

__device__ inline void store_tile_D_f16_T(v8f d, _Float16* p, int ld) {
  int lane = threadIdx.x & 31;
  int col  = lane & 15;
  int row0 = (lane >> 4) * 8;
#pragma unroll
  for (int r = 0; r < 8; ++r) p[(size_t)col * ld + row0 + r] = (_Float16)d[r];
}

__device__ inline float rowmax16(float v) {
#pragma unroll
  for (int off = 1; off < 16; off <<= 1) v = fmaxf(v, __shfl_xor(v, off, 32));
  return v;
}
__device__ inline float rowsum16(float v) {
#pragma unroll
  for (int off = 1; off < 16; off <<= 1) v += __shfl_xor(v, off, 32);
  return v;
}

// ---------------------------------------------------------------------------
// Kernel 0: weight fp32 -> f16 conversion (row-major preserved).
// ---------------------------------------------------------------------------
__global__ void k_wcvt(const float* __restrict__ Wq, const float* __restrict__ Wk,
                       const float* __restrict__ Wv, const float* __restrict__ W0,
                       _Float16* Wq16, _Float16* Wk16, _Float16* Wv16, _Float16* W016) {
  int idx   = blockIdx.x * blockDim.x + threadIdx.x;   // 4 * 65536 elements
  int which = idx >> 16;
  int t     = idx & 65535;
  const float* src = (which == 0) ? Wq : (which == 1) ? Wk : (which == 2) ? Wv : W0;
  _Float16*    dst = (which == 0) ? Wq16 : (which == 1) ? Wk16 : (which == 2) ? Wv16 : W016;
  dst[t] = (_Float16)src[t];
}

// ---------------------------------------------------------------------------
// Kernel 1: xT[b,n,c] = x[b,c,n] * inv_layer (fp32 -> f16)
// ---------------------------------------------------------------------------
__global__ void k_xT(const float* __restrict__ x, _Float16* __restrict__ xT) {
  int idx = blockIdx.x * blockDim.x + threadIdx.x;     // B*C*N
  int n = idx & (kN - 1);
  int c = (idx >> 10) & (kC - 1);
  int b = idx >> 17;
  xT[((size_t)b * kN + n) * kC + c] = (_Float16)(x[idx] * kInvLayer);
}

// ---------------------------------------------------------------------------
// Kernel 2: QKV projection via WMMA.
//   QT/KT stored (B,H,N,DK) row-major; V stored (B,H,DK,N).
// grid = B * (N/16) * 3 blocks, 128 threads (4 waves x 128 out-cols).
// ---------------------------------------------------------------------------
__global__ __launch_bounds__(128) void k_qkv(
    const _Float16* __restrict__ xT,
    const _Float16* __restrict__ Wq16, const _Float16* __restrict__ Wk16,
    const _Float16* __restrict__ Wv16,
    _Float16* __restrict__ QT, _Float16* __restrict__ KT, _Float16* __restrict__ Vh) {
  int m     = blockIdx.x;
  int which = m % 3;
  int itile = (m / 3) & 63;
  int b     = m / (3 * 64);
  int wave  = threadIdx.x >> 5;
  int i0    = itile * 16;
  int r0    = wave * 128;

  const _Float16* W  = (which == 0) ? Wq16 : (which == 1) ? Wk16 : Wv16;
  const _Float16* Ab = xT + ((size_t)b * kN + i0) * kC;

  v16h a[4];
#pragma unroll
  for (int cc = 0; cc < 4; ++cc) a[cc] = load_tile_A(Ab + cc * 32, kC);

#pragma unroll
  for (int tt = 0; tt < 8; ++tt) {
    const _Float16* Wb = W + (size_t)(r0 + tt * 16) * kC;
    v16h b0 = load_tile_A(Wb,      kC);
    v16h b1 = load_tile_A(Wb + 32, kC);
    v16h b2 = load_tile_A(Wb + 64, kC);
    v16h b3 = load_tile_A(Wb + 96, kC);
    v8f acc = (v8f){0, 0, 0, 0, 0, 0, 0, 0};
    acc = wmma16(a[0], b0, acc);
    acc = wmma16(a[1], b1, acc);
    acc = wmma16(a[2], b2, acc);
    acc = wmma16(a[3], b3, acc);

    int r  = r0 + tt * 16;
    int h  = r >> 6;
    int d0 = r & 63;
    if (which == 2) {
      store_tile_D_f16_T(acc, Vh + (((size_t)(b * kH + h) * kDK + d0) * kN + i0), kN);
    } else {
      _Float16* dst = (which == 0) ? QT : KT;
      store_tile_D_f16(acc, dst + (((size_t)(b * kH + h) * kN + i0) * kDK + d0), kDK);
    }
  }
}

// ---------------------------------------------------------------------------
// Kernel 3: flash attention with on-the-fly iRPE bias.
// grid = B*H*(N/64) blocks, 128 threads (4 waves x 16 query rows).
// K/V chunks (32 keys) staged cooperatively into LDS with async
// global->LDS copies (ASYNCcnt), double-buffered; B-tiles then come from LDS.
// ---------------------------------------------------------------------------
__global__ __launch_bounds__(128) void k_attn(
    const _Float16* __restrict__ QT, const _Float16* __restrict__ KT,
    const _Float16* __restrict__ Vh, const float* __restrict__ R,
    _Float16* __restrict__ attT) {
  __shared__ float    R_s[kL * kL];        // 4 KB bias table for this head
  __shared__ _Float16 Ks[2][32 * kDK];     // 2 x 4 KB : 32 keys x 64 d
  __shared__ _Float16 Vs[2][kDK * 32];     // 2 x 4 KB : 64 d x 32 keys
  __shared__ _Float16 P_s[4 * 16 * 32];    // 4 KB per-wave P re-striping

  int qt = blockIdx.x & 15;
  int bh = blockIdx.x >> 4;
  int h  = bh & 7;
  int b  = bh >> 3;
  int t  = threadIdx.x;

  size_t headoff = (size_t)(b * kH + h) * kN * kDK;
  const _Float16* Kg0 = KT + headoff;
  const _Float16* Vg0 = Vh + headoff;

  // stage one 32-key chunk (K: 4KB contiguous; V: 64 rows x 64B) into buf
  auto stage = [&](int jc, int buf) {
#pragma unroll
    for (int u = 0; u < 2; ++u) {
      int c = t + u * 128;               // 256 x 16B chunks each for K and V
      async_copy16(Kg0 + (size_t)jc * 32 * kDK + c * 8, &Ks[buf][c * 8]);
      int d = c >> 2, o = (c & 3) * 8;
      async_copy16(Vg0 + (size_t)d * kN + jc * 32 + o, &Vs[buf][d * 32 + o]);
    }
  };

  for (int u = t; u < kL * kL; u += 128) R_s[u] = R[(size_t)h * kL * kL + u];
  stage(0, 0);

  int wave = t >> 5;
  int lane = t & 31;
  int half = lane >> 4;
  int colA = lane & 15;
  int i0   = qt * 64 + wave * 16;
  int rowb = i0 + half * 8;

  const _Float16* Qb = QT + headoff + (size_t)i0 * kDK;
  v16h aq0 = load_tile_A(Qb, kDK);
  v16h aq1 = load_tile_A(Qb + 32, kDK);

  v8f o0 = (v8f){0,0,0,0,0,0,0,0}, o1 = o0, o2 = o0, o3 = o0;
  float m8[8], l8[8];
#pragma unroll
  for (int r = 0; r < 8; ++r) { m8[r] = -3.0e38f; l8[r] = 0.0f; }

  _Float16* Pw = P_s + wave * 512;

  for (int jc = 0; jc < kN / 32; ++jc) {
    int buf = jc & 1;
    async_wait0();        // this wave's staged chunk has landed in LDS
    __syncthreads();      // everyone's chunk landed; prev buffer fully read
    if (jc + 1 < kN / 32) stage(jc + 1, buf ^ 1);

    // ---- S tiles (16q x 32k) from LDS K, 4 WMMAs ----
    const _Float16* Kb = &Ks[buf][0];
    v8f s0 = (v8f){0,0,0,0,0,0,0,0}, s1 = s0;
    {
      v16h bk0 = load_tile_A(Kb,          kDK);
      v16h bk1 = load_tile_A(Kb + 32,     kDK);
      v16h bk2 = load_tile_A(Kb + 16*kDK,      kDK);
      v16h bk3 = load_tile_A(Kb + 16*kDK + 32, kDK);
      s0 = wmma16(aq0, bk0, s0);
      s0 = wmma16(aq1, bk1, s0);
      s1 = wmma16(aq0, bk2, s1);
      s1 = wmma16(aq1, bk3, s1);
    }
    int j0 = jc * 32;
    // ---- iRPE bias + scale ----
#pragma unroll
    for (int r = 0; r < 8; ++r) {
      int i  = rowb + r;
      int i1 = i >> 5, i2 = i & 31;
      int jA = j0 + colA;
      int jB = jA + 16;
      float bA = R_s[(((i1 - (jA >> 5)) & 31) << 5) | ((i2 - (jA & 31)) & 31)];
      float bB = R_s[(((i1 - (jB >> 5)) & 31) << 5) | ((i2 - (jB & 31)) & 31)];
      s0[r] = (s0[r] + bA) * kInvSqrtDk;
      s1[r] = (s1[r] + bB) * kInvSqrtDk;
    }
    // ---- online softmax (stats per lane-half x r) ----
#pragma unroll
    for (int r = 0; r < 8; ++r) {
      float mx    = rowmax16(fmaxf(s0[r], s1[r]));
      float m_new = fmaxf(m8[r], mx);
      float corr  = __expf(m8[r] - m_new);
      float p0    = __expf(s0[r] - m_new);
      float p1    = __expf(s1[r] - m_new);
      l8[r] = l8[r] * corr + rowsum16(p0 + p1);
      m8[r] = m_new;
      o0[r] *= corr; o1[r] *= corr; o2[r] *= corr; o3[r] *= corr;
      int rl = (half * 8 + r) << 5;           // C-layout -> LDS row-major
      Pw[rl + colA]      = (_Float16)p0;
      Pw[rl + 16 + colA] = (_Float16)p1;
    }
    // per-wave LDS RAW: drain DScnt before re-striping P into A-layout
    asm volatile("s_wait_dscnt 0" ::: "memory");
    v16h pa = load_tile_A(Pw, 32);
    // ---- O += P @ V from LDS (B columns row-contiguous) ----
    const _Float16* Vb = &Vs[buf][0];
    o0 = wmma16(pa, load_tile_A(Vb,           32), o0);
    o1 = wmma16(pa, load_tile_A(Vb + 16 * 32, 32), o1);
    o2 = wmma16(pa, load_tile_A(Vb + 32 * 32, 32), o2);
    o3 = wmma16(pa, load_tile_A(Vb + 48 * 32, 32), o3);
  }

#pragma unroll
  for (int r = 0; r < 8; ++r) {
    float inv = 1.0f / l8[r];
    o0[r] *= inv; o1[r] *= inv; o2[r] *= inv; o3[r] *= inv;
  }
  _Float16* Ob = attT + ((size_t)b * kN + i0) * kHD + h * kDK;
  store_tile_D_f16(o0, Ob,      kHD);
  store_tile_D_f16(o1, Ob + 16, kHD);
  store_tile_D_f16(o2, Ob + 32, kHD);
  store_tile_D_f16(o3, Ob + 48, kHD);
}

// ---------------------------------------------------------------------------
// Kernel 4: output projection + residual.
//   out[b,c,i] = sum_hd attT[b,i,hd] * W0[c,hd] + x[b,c,i]
// ---------------------------------------------------------------------------
__global__ __launch_bounds__(128) void k_out(
    const _Float16* __restrict__ attT, const _Float16* __restrict__ W016,
    const float* __restrict__ x, float* __restrict__ out) {
  int itile = blockIdx.x & 63;
  int b     = blockIdx.x >> 6;
  int wave  = threadIdx.x >> 5;
  int lane  = threadIdx.x & 31;
  int i0    = itile * 16;
  int c0    = wave * 32;

  v8f acc0 = (v8f){0,0,0,0,0,0,0,0}, acc1 = acc0;
  const _Float16* Ab = attT + ((size_t)b * kN + i0) * kHD;
#pragma unroll 4
  for (int k = 0; k < kHD; k += 32) {
    v16h a  = load_tile_A(Ab + k, kHD);
    v16h b0 = load_tile_A(W016 + (size_t)c0 * kHD + k, kHD);
    v16h b1 = load_tile_A(W016 + (size_t)(c0 + 16) * kHD + k, kHD);
    acc0 = wmma16(a, b0, acc0);
    acc1 = wmma16(a, b1, acc1);
  }

  int colA = lane & 15;
  int row0 = (lane >> 4) * 8;
#pragma unroll
  for (int r = 0; r < 8; ++r) {
    int i = i0 + row0 + r;
    size_t idx0 = ((size_t)(b * kC) + c0 + colA) * kN + i;
    size_t idx1 = ((size_t)(b * kC) + c0 + 16 + colA) * kN + i;
    out[idx0] = acc0[r] + x[idx0];
    out[idx1] = acc1[r] + x[idx1];
  }
}

// ---------------------------------------------------------------------------
// Host-side orchestration (graph-capture safe: only kernel launches on stream)
// ---------------------------------------------------------------------------
extern "C" void kernel_launch(void* const* d_in, const int* in_sizes, int n_in,
                              void* d_out, int out_size, void* d_ws, size_t ws_size,
                              hipStream_t stream) {
  (void)in_sizes; (void)n_in; (void)out_size; (void)ws_size;
  const float* x  = (const float*)d_in[0];
  const float* Wq = (const float*)d_in[1];
  const float* Wk = (const float*)d_in[2];
  const float* Wv = (const float*)d_in[3];
  const float* R  = (const float*)d_in[4];
  const float* W0 = (const float*)d_in[5];
  float* out = (float*)d_out;

  _Float16* ws = (_Float16*)d_ws;
  size_t off = 0;
  _Float16* xT   = ws + off; off += (size_t)kB * kN * kC;
  _Float16* Wq16 = ws + off; off += (size_t)kHD * kC;
  _Float16* Wk16 = ws + off; off += (size_t)kHD * kC;
  _Float16* Wv16 = ws + off; off += (size_t)kHD * kC;
  _Float16* W016 = ws + off; off += (size_t)kC * kHD;
  _Float16* QT   = ws + off; off += (size_t)kB * kH * kN * kDK;
  _Float16* KT   = ws + off; off += (size_t)kB * kH * kN * kDK;
  _Float16* Vh   = ws + off; off += (size_t)kB * kH * kDK * kN;
  _Float16* attT = ws + off; off += (size_t)kB * kN * kHD;

  k_wcvt<<<(4 * kHD * kC) / 256, 256, 0, stream>>>(Wq, Wk, Wv, W0, Wq16, Wk16, Wv16, W016);
  k_xT<<<(kB * kC * kN) / 256, 256, 0, stream>>>(x, xT);
  k_qkv<<<kB * (kN / 16) * 3, 128, 0, stream>>>(xT, Wq16, Wk16, Wv16, QT, KT, Vh);
  k_attn<<<kB * kH * (kN / 64), 128, 0, stream>>>(QT, KT, Vh, R, attT);
  k_out<<<kB * (kN / 16), 128, 0, stream>>>(attT, W016, x, out);
}
